// MultiHeadAttention_19361712570445
// MI455X (gfx1250) — compile-verified, matrix operand order checked by reading, not run on hardware
//
#include <hip/hip_runtime.h>

// ---------------------------------------------------------------------------
// MHA forward for MI455X (gfx1250): bf16 WMMA GEMMs + flash attention,
// with Tensor Data Mover (TDM) async tile staging into LDS.
// B=2, S=2048, D=1024, H=16, HD=64.
// ---------------------------------------------------------------------------

#define BB 2
#define SS 2048
#define DD 1024
#define HH 16
#define HD 64
#define MTOT (BB * SS)   // 4096
#define NEGV (-1e9f)

typedef __attribute__((ext_vector_type(16))) __bf16       v16bf;
typedef __attribute__((ext_vector_type(8)))  float        v8f;
typedef __attribute__((ext_vector_type(4)))  unsigned int v4u;
typedef __attribute__((ext_vector_type(2)))  unsigned int v2u;
typedef __attribute__((ext_vector_type(4)))  unsigned int u32x4;
typedef __attribute__((ext_vector_type(8)))  int          i32x8;
typedef __attribute__((ext_vector_type(4)))  int          i32x4;

#ifndef __has_builtin
#define __has_builtin(x) 0
#endif
#if __has_builtin(__builtin_amdgcn_tensor_load_to_lds) && \
    __has_builtin(__builtin_amdgcn_s_wait_tensorcnt)
#define HAVE_TDM 1
#else
#define HAVE_TDM 0
#endif

union FragB16 { v4u q[2]; v16bf h; };   // 32 bytes: 16 bf16 elems per lane

__device__ __forceinline__ unsigned short f32_to_bf16(float f) {
  unsigned int u = __builtin_bit_cast(unsigned int, f);
  unsigned int r = u + 0x7FFFu + ((u >> 16) & 1u);   // round-to-nearest-even
  return (unsigned short)(r >> 16);
}

#if HAVE_TDM
// LDS offset of a __shared__ object: generic LDS addr low 32 bits (ISA 10.2).
__device__ __forceinline__ unsigned lds_off(const void* p) {
  return (unsigned)(unsigned long long)p;
}
// Issue a 2-D TDM tile load: bf16 elements, row-padded LDS destination.
// tile/tensor dims in elements; stride = row stride in elements.
// pad_iv: pad_interval code (interval = 8<<pad_iv bytes); pad_am: code (=(v+1) dwords).
__device__ __forceinline__ void tdm_load_2d(unsigned lds_addr, const void* gp,
                                            unsigned td0, unsigned td1,
                                            unsigned tl0, unsigned tl1,
                                            unsigned stride,
                                            unsigned pad_iv, unsigned pad_am) {
  unsigned long long ga = (unsigned long long)gp;
  u32x4 g0;
  g0.x = 1u;                                           // count=1, user mode
  g0.y = lds_addr;                                     // LDS byte address
  g0.z = (unsigned)ga;                                 // global_addr[31:0]
  g0.w = ((unsigned)(ga >> 32) & 0x01FFFFFFu) | (2u << 30);  // addr[56:32]|type=2
  i32x8 g1;
  g1[0] = (int)((1u << 16) | (1u << 20) | (pad_iv << 22) | (pad_am << 25));
  g1[1] = (int)(td0 << 16);                            // tensor_dim0 lo16 @48
  g1[2] = (int)((td0 >> 16) | (td1 << 16));            // dim0 hi | dim1 lo
  g1[3] = (int)((td1 >> 16) | (tl0 << 16));            // dim1 hi | tile_dim0
  g1[4] = (int)(tl1 & 0xFFFFu);                        // tile_dim1 (tile_dim2=0)
  g1[5] = (int)stride;                                 // tensor_dim0_stride lo32
  g1[6] = 0;
  g1[7] = 0;
  i32x4 z4 = {0, 0, 0, 0};
  i32x8 z8 = {0, 0, 0, 0, 0, 0, 0, 0};
  // 6-arg form (clang-23 / therock-10.0): groups 2,3 + extra group, then cpol.
  __builtin_amdgcn_tensor_load_to_lds(g0, g1, z4, z4, z8, 0);
}
#endif

// ---------------------------------------------------------------------------
// fp32 -> bf16 conversion, 4 elements/thread (activations)
// ---------------------------------------------------------------------------
__global__ void cvt_bf16x4(const float4* __restrict__ src,
                           v2u* __restrict__ dst, int n4) {
  int i = blockIdx.x * blockDim.x + threadIdx.x;
  if (i < n4) {
    float4 f = src[i];
    v2u o;
    o.x = (unsigned int)f32_to_bf16(f.x) | ((unsigned int)f32_to_bf16(f.y) << 16);
    o.y = (unsigned int)f32_to_bf16(f.z) | ((unsigned int)f32_to_bf16(f.w) << 16);
    dst[i] = o;
  }
}

// ---------------------------------------------------------------------------
// fp32 W[K][N] -> bf16 Wt[N][K] transpose-convert, 32x32 LDS tile.
// ---------------------------------------------------------------------------
__global__ __launch_bounds__(256)
void cvt_transpose_bf16(const float* __restrict__ src,
                        unsigned short* __restrict__ dst, int K, int N) {
  __shared__ unsigned short tile[32][33];
  const int tx = threadIdx.x & 31, ty = threadIdx.x >> 5;
  const int k0 = blockIdx.y * 32, n0 = blockIdx.x * 32;
#pragma unroll
  for (int p = 0; p < 4; ++p) {
    int kr = ty + p * 8;
    tile[kr][tx] = f32_to_bf16(src[(size_t)(k0 + kr) * N + n0 + tx]);
  }
  __syncthreads();
#pragma unroll
  for (int p = 0; p < 4; ++p) {
    int nr = ty + p * 8;
    dst[(size_t)(n0 + nr) * K + k0 + tx] = tile[tx][nr];
  }
}

// ---------------------------------------------------------------------------
// bf16 GEMM: C[M,N] = A[M,K] @ Wt[N,K]^T + bias, f32 accumulate via WMMA.
// WG = 256 threads (8 waves), WG tile 128x256, wave tile 64x64, K-step 32.
// TDM path: wave 0 DMAs A/B k-tiles into double-buffered LDS (row pad 16B ->
// stride-40 layout), TENSORcnt-synchronized; DMA for tile k+1 overlaps the
// 16 WMMAs of tile k. Fallback: register-prefetch staging.
// Per-lane fragment layout (16-bit), g = lane>>4:
//   element e -> k = (e&7) | (g<<3) | ((e>>3)<<4) => chunks [8g..+7],[16+8g..+7]
// ---------------------------------------------------------------------------
__global__ __launch_bounds__(256)
void gemm_bf16(const unsigned short* __restrict__ A,
               const unsigned short* __restrict__ Wt,
               const float* __restrict__ bias, float scale,
               unsigned short* __restrict__ Obf,
               unsigned short* __restrict__ ObfT,
               float* __restrict__ Of32,
               int M, int N, int K) {
#if HAVE_TDM
  __shared__ __align__(16) unsigned short As[2][128][40];
  __shared__ __align__(16) unsigned short Bs[2][256][40];
#else
  __shared__ __align__(16) unsigned short As[1][128][40];
  __shared__ __align__(16) unsigned short Bs[1][256][40];
#endif

  const int t    = threadIdx.x;
  const int wave = t >> 5;
  const int lane = t & 31;
  const int g    = lane >> 4;
  const int r    = lane & 15;
  const int wm   = (wave >> 2) * 64;
  const int wn   = (wave & 3) * 64;
  const int bm   = blockIdx.y * 128;
  const int bn   = blockIdx.x * 256;
  const int nk   = K >> 5;

  v8f acc[4][4];
#pragma unroll
  for (int i = 0; i < 4; ++i)
#pragma unroll
    for (int j = 0; j < 4; ++j)
      acc[i][j] = v8f{0.f, 0.f, 0.f, 0.f, 0.f, 0.f, 0.f, 0.f};

#if HAVE_TDM
  if (wave == 0) {   // prologue: DMA k-tile 0 into buffer 0
    tdm_load_2d(lds_off(&As[0][0][0]), A  + (size_t)bm * K, 32, 128, 32, 128, K, 3, 3);
    tdm_load_2d(lds_off(&Bs[0][0][0]), Wt + (size_t)bn * K, 32, 256, 32, 256, K, 3, 3);
  }
#else
  const int srow = t >> 2;
  const int sc8  = (t & 3) << 3;
  const unsigned short* pA = A  + (size_t)(bm + srow) * K + sc8;
  const unsigned short* pB = Wt + (size_t)(bn + srow) * K + sc8;
  v4u ra[2], rb[4];
#pragma unroll
  for (int p = 0; p < 2; ++p) ra[p] = *(const v4u*)(pA + (size_t)(64 * p) * K);
#pragma unroll
  for (int p = 0; p < 4; ++p) rb[p] = *(const v4u*)(pB + (size_t)(64 * p) * K);
#endif

  for (int kt = 0; kt < nk; ++kt) {
#if HAVE_TDM
    __syncthreads();   // all waves done reading buffer (kt+1)&1 (iter kt-1)
    if (wave == 0) {
      if (kt + 1 < nk) {   // DMA next tile; overlaps this tile's WMMAs
        const size_t k0 = (size_t)(kt + 1) << 5;
        const int nb = (kt + 1) & 1;
        tdm_load_2d(lds_off(&As[nb][0][0]), A  + (size_t)bm * K + k0,
                    32, 128, 32, 128, K, 3, 3);
        tdm_load_2d(lds_off(&Bs[nb][0][0]), Wt + (size_t)bn * K + k0,
                    32, 256, 32, 256, K, 3, 3);
        __builtin_amdgcn_s_wait_tensorcnt(2);   // tile kt (2 ops) complete
      } else {
        __builtin_amdgcn_s_wait_tensorcnt(0);
      }
    }
    __syncthreads();   // buffer kt&1 published
    const int buf = kt & 1;
#else
    __syncthreads();
#pragma unroll
    for (int p = 0; p < 2; ++p) *(v4u*)(&As[0][srow + 64 * p][sc8]) = ra[p];
#pragma unroll
    for (int p = 0; p < 4; ++p) *(v4u*)(&Bs[0][srow + 64 * p][sc8]) = rb[p];
    __syncthreads();
    if (kt + 1 < nk) {
      const int k0 = (kt + 1) << 5;
#pragma unroll
      for (int p = 0; p < 2; ++p)
        ra[p] = *(const v4u*)(pA + (size_t)(64 * p) * K + k0);
#pragma unroll
      for (int p = 0; p < 4; ++p)
        rb[p] = *(const v4u*)(pB + (size_t)(64 * p) * K + k0);
    }
    const int buf = 0;
#endif

    FragB16 af[4], bf[4];
#pragma unroll
    for (int i = 0; i < 4; ++i) {
      const unsigned short* p = &As[buf][wm + i * 16 + r][g * 8];
      af[i].q[0] = *(const v4u*)p;
      af[i].q[1] = *(const v4u*)(p + 16);
    }
#pragma unroll
    for (int j = 0; j < 4; ++j) {
      const unsigned short* p = &Bs[buf][wn + j * 16 + r][g * 8];
      bf[j].q[0] = *(const v4u*)p;
      bf[j].q[1] = *(const v4u*)(p + 16);
    }
#pragma unroll
    for (int i = 0; i < 4; ++i)
#pragma unroll
      for (int j = 0; j < 4; ++j)
        acc[i][j] = __builtin_amdgcn_wmma_f32_16x16x32_bf16(
            false, af[i].h, false, bf[j].h, (short)0, acc[i][j], false, false);
  }

  // epilogue: C layout -> element d of lane = (row d + 8g, col r)
#pragma unroll
  for (int j = 0; j < 4; ++j) {
    const int col = bn + wn + j * 16 + r;
    const float bj = bias[col];
#pragma unroll
    for (int i = 0; i < 4; ++i) {
      const int row0 = bm + wm + i * 16 + g * 8;
      float vv[8];
#pragma unroll
      for (int d = 0; d < 8; ++d) vv[d] = (acc[i][j][d] + bj) * scale;
      if (Of32) {
#pragma unroll
        for (int d = 0; d < 8; ++d) Of32[(size_t)(row0 + d) * N + col] = vv[d];
      }
      if (Obf) {
#pragma unroll
        for (int d = 0; d < 8; ++d) Obf[(size_t)(row0 + d) * N + col] = f32_to_bf16(vv[d]);
      }
      if (ObfT) {   // 8 consecutive rows, fixed col -> one 16B store
        v4u pk;
        pk.x = (unsigned int)f32_to_bf16(vv[0]) | ((unsigned int)f32_to_bf16(vv[1]) << 16);
        pk.y = (unsigned int)f32_to_bf16(vv[2]) | ((unsigned int)f32_to_bf16(vv[3]) << 16);
        pk.z = (unsigned int)f32_to_bf16(vv[4]) | ((unsigned int)f32_to_bf16(vv[5]) << 16);
        pk.w = (unsigned int)f32_to_bf16(vv[6]) | ((unsigned int)f32_to_bf16(vv[7]) << 16);
        *(v4u*)(ObfT + (size_t)col * M + row0) = pk;
      }
    }
  }
}

// ---------------------------------------------------------------------------
// Flash attention. Grid: (S/64, H, B). WG = 128 threads (4 waves), wave owns
// 16 q-rows. Q pre-scaled by 1/sqrt(HD). Causal mask computed analytically.
// V arrives pre-transposed (VpT[dim][b*S+s]); V tiles DMAed by TDM into
// double-buffered LDS (or b128 staging fallback). kv tile = 64 keys/iter.
// ---------------------------------------------------------------------------
__global__ __launch_bounds__(128)
void flash_attn(const unsigned short* __restrict__ Q,
                const unsigned short* __restrict__ Kk,
                const unsigned short* __restrict__ VpT,  // [D][MTOT]
                const float* __restrict__ kpm,           // (B,S)
                unsigned short* __restrict__ Ctx) {      // (B*S, D) bf16
#if HAVE_TDM
  __shared__ __align__(16) unsigned short Vt[2][64][72];    // [buf][dim][key]
#else
  __shared__ __align__(16) unsigned short Vt[1][64][72];
#endif
  __shared__ __align__(16) unsigned short Ps[4][16][72];    // per-wave P

  const int t    = threadIdx.x;
  const int wave = t >> 5;
  const int lane = t & 31;
  const int g    = lane >> 4;
  const int r    = lane & 15;
  const int qblk = blockIdx.x * 64;
  const int h    = blockIdx.y;
  const int b    = blockIdx.z;
  const int qrow0 = qblk + wave * 16;
  const unsigned short* pVbase = VpT + (size_t)(h * HD) * MTOT + b * SS;

#if !HAVE_TDM
  const int vrow = t >> 3;
  const int vc8  = (t & 7) << 3;
#endif

  // Q A-fragments for the two HD k-steps, loaded once.
  FragB16 aq[2];
#pragma unroll
  for (int kk = 0; kk < 2; ++kk) {
    const unsigned short* p =
        Q + (size_t)(b * SS + qrow0 + r) * DD + h * HD + kk * 32 + g * 8;
    aq[kk].q[0] = *(const v4u*)p;
    aq[kk].q[1] = *(const v4u*)(p + 16);
  }

  float m_run[8], l_run[8];
  v8f acc_o[4];
#pragma unroll
  for (int d = 0; d < 8; ++d) { m_run[d] = -1e30f; l_run[d] = 0.f; }
#pragma unroll
  for (int jo = 0; jo < 4; ++jo)
    acc_o[jo] = v8f{0.f, 0.f, 0.f, 0.f, 0.f, 0.f, 0.f, 0.f};

  const int ntiles = blockIdx.x + 1;                   // causal bound

#if HAVE_TDM
  if (wave == 0)   // prologue: DMA V-tile 0 (64 dims x 64 keys) into buf 0
    tdm_load_2d(lds_off(&Vt[0][0][0]), pVbase, 64, 64, 64, 64, MTOT, 4, 3);
#endif

  for (int it = 0; it < ntiles; ++it) {
    const int kv0 = it * 64;
#if HAVE_TDM
    __syncthreads();   // prior PV reads of buf (it+1)&1 complete
    if (wave == 0 && it + 1 < ntiles)
      tdm_load_2d(lds_off(&Vt[(it + 1) & 1][0][0]), pVbase + kv0 + 64,
                  64, 64, 64, 64, MTOT, 4, 3);
    const int vb = it & 1;
#else
    v4u rv[4];
#pragma unroll
    for (int p = 0; p < 4; ++p)
      rv[p] = *(const v4u*)(pVbase + (size_t)(vrow + 16 * p) * MTOT + kv0 + vc8);
    __syncthreads();
#pragma unroll
    for (int p = 0; p < 4; ++p)
      *(v4u*)(&Vt[0][vrow + 16 * p][vc8]) = rv[p];
    const int vb = 0;
#endif

    // ---- S = Q K^T (+ masks): batch all K-fragment loads, then 8 WMMAs ----
    FragB16 bk[4][2];
#pragma unroll
    for (int jj = 0; jj < 4; ++jj) {
      const int key = kv0 + jj * 16 + r;
      const unsigned short* p =
          Kk + (size_t)(b * SS + key) * DD + h * HD + g * 8;
      bk[jj][0].q[0] = *(const v4u*)p;
      bk[jj][0].q[1] = *(const v4u*)(p + 16);
      bk[jj][1].q[0] = *(const v4u*)(p + 32);
      bk[jj][1].q[1] = *(const v4u*)(p + 48);
    }
    float s[4][8];
#pragma unroll
    for (int jj = 0; jj < 4; ++jj) {
      v8f sa = v8f{0.f, 0.f, 0.f, 0.f, 0.f, 0.f, 0.f, 0.f};
      sa = __builtin_amdgcn_wmma_f32_16x16x32_bf16(
          false, aq[0].h, false, bk[jj][0].h, (short)0, sa, false, false);
      sa = __builtin_amdgcn_wmma_f32_16x16x32_bf16(
          false, aq[1].h, false, bk[jj][1].h, (short)0, sa, false, false);
      const int key = kv0 + jj * 16 + r;
      const float kp = kpm[b * SS + key];
#pragma unroll
      for (int d = 0; d < 8; ++d) {
        int qr = qrow0 + d + 8 * g;
        float cm = (key <= qr) ? 0.f : NEGV;           // analytic tril mask
        s[jj][d] = sa[d] + kp + cm;
      }
    }

    // ---- online softmax ----
    float alpha[8];
#pragma unroll
    for (int d = 0; d < 8; ++d) {
      float m = fmaxf(fmaxf(s[0][d], s[1][d]), fmaxf(s[2][d], s[3][d]));
      m = fmaxf(m, __shfl_xor(m, 1));
      m = fmaxf(m, __shfl_xor(m, 2));
      m = fmaxf(m, __shfl_xor(m, 4));
      m = fmaxf(m, __shfl_xor(m, 8));
      float mn = fmaxf(m_run[d], m);
      alpha[d] = __expf(m_run[d] - mn);
      m_run[d] = mn;
      float rs = 0.f;
#pragma unroll
      for (int jj = 0; jj < 4; ++jj) {
        float p = __expf(s[jj][d] - mn);
        rs += p;
        Ps[wave][d + 8 * g][jj * 16 + r] = f32_to_bf16(p);
      }
      rs += __shfl_xor(rs, 1);
      rs += __shfl_xor(rs, 2);
      rs += __shfl_xor(rs, 4);
      rs += __shfl_xor(rs, 8);
      l_run[d] = l_run[d] * alpha[d] + rs;
    }
#pragma unroll
    for (int jo = 0; jo < 4; ++jo)
#pragma unroll
      for (int d = 0; d < 8; ++d) acc_o[jo][d] *= alpha[d];

#if HAVE_TDM
    if (wave == 0) {   // tile `it` done (next tile may stay in flight)
      if (it + 1 < ntiles) __builtin_amdgcn_s_wait_tensorcnt(1);
      else                 __builtin_amdgcn_s_wait_tensorcnt(0);
    }
#endif
    __syncthreads();                                   // Vt[vb] ready

    // ---- O += P @ V ----
#pragma unroll
    for (int kk = 0; kk < 2; ++kk) {
      FragB16 ap;
      const unsigned short* pp = &Ps[wave][r][kk * 32 + g * 8];
      ap.q[0] = *(const v4u*)pp;
      ap.q[1] = *(const v4u*)(pp + 16);
#pragma unroll
      for (int jo = 0; jo < 4; ++jo) {
        FragB16 bv;
        const unsigned short* p = &Vt[vb][jo * 16 + r][kk * 32 + g * 8];
        bv.q[0] = *(const v4u*)p;
        bv.q[1] = *(const v4u*)(p + 16);
        acc_o[jo] = __builtin_amdgcn_wmma_f32_16x16x32_bf16(
            false, ap.h, false, bv.h, (short)0, acc_o[jo], false, false);
      }
    }
  }

  // ---- normalize and write ctx (bf16) ----
#pragma unroll
  for (int jo = 0; jo < 4; ++jo)
#pragma unroll
    for (int d = 0; d < 8; ++d) {
      int qr = qrow0 + d + 8 * g;
      float o = acc_o[jo][d] / l_run[d];
      Ctx[(size_t)(b * SS + qr) * DD + h * HD + jo * 16 + r] = f32_to_bf16(o);
    }
}

// ---------------------------------------------------------------------------
// Host launch
// ---------------------------------------------------------------------------
extern "C" void kernel_launch(void* const* d_in, const int* in_sizes, int n_in,
                              void* d_out, int out_size, void* d_ws,
                              size_t ws_size, hipStream_t stream) {
  const float* q   = (const float*)d_in[0];
  const float* k   = (const float*)d_in[1];
  const float* v   = (const float*)d_in[2];
  const float* Wq  = (const float*)d_in[3];
  const float* bq  = (const float*)d_in[4];
  const float* Wk  = (const float*)d_in[5];
  const float* bk  = (const float*)d_in[6];
  const float* Wv  = (const float*)d_in[7];
  const float* bv  = (const float*)d_in[8];
  const float* Wo  = (const float*)d_in[9];
  const float* bo  = (const float*)d_in[10];
  // d_in[11] = attention_mask (causal tril) -> computed analytically on-chip
  const float* kpm = (const float*)d_in[12];

  const size_t MB = 1024 * 1024;
  char* ws = (char*)d_ws;
  unsigned short* Xq  = (unsigned short*)(ws + 0 * MB);
  unsigned short* Xk  = (unsigned short*)(ws + 8 * MB);
  unsigned short* Xv  = (unsigned short*)(ws + 16 * MB);
  unsigned short* Wqt = (unsigned short*)(ws + 24 * MB);  // [N][K]
  unsigned short* Wkt = (unsigned short*)(ws + 26 * MB);
  unsigned short* Wvt = (unsigned short*)(ws + 28 * MB);
  unsigned short* Wot = (unsigned short*)(ws + 30 * MB);
  unsigned short* Qp  = (unsigned short*)(ws + 32 * MB);
  unsigned short* Kp  = (unsigned short*)(ws + 40 * MB);
  unsigned short* VpT = (unsigned short*)(ws + 48 * MB);  // [D][MTOT]
  unsigned short* Ctx = (unsigned short*)(ws + 56 * MB);

  const int nAct = MTOT * DD;
  auto cvt = [&](const float* src, unsigned short* dst, int n) {
    int n4 = n / 4;
    cvt_bf16x4<<<(n4 + 255) / 256, 256, 0, stream>>>((const float4*)src,
                                                     (v2u*)dst, n4);
  };
  cvt(q, Xq, nAct);  cvt(k, Xk, nAct);  cvt(v, Xv, nAct);

  dim3 gt(DD / 32, DD / 32);
  cvt_transpose_bf16<<<gt, 256, 0, stream>>>(Wq, Wqt, DD, DD);
  cvt_transpose_bf16<<<gt, 256, 0, stream>>>(Wk, Wkt, DD, DD);
  cvt_transpose_bf16<<<gt, 256, 0, stream>>>(Wv, Wvt, DD, DD);
  cvt_transpose_bf16<<<gt, 256, 0, stream>>>(Wo, Wot, DD, DD);

  dim3 gg(DD / 256, MTOT / 128);   // (4, 32)
  const float isq = 0.125f;        // 1/sqrt(64)
  gemm_bf16<<<gg, 256, 0, stream>>>(Xq, Wqt, bq, isq, Qp, nullptr, nullptr,
                                    MTOT, DD, DD);
  gemm_bf16<<<gg, 256, 0, stream>>>(Xk, Wkt, bk, 1.f, Kp, nullptr, nullptr,
                                    MTOT, DD, DD);
  gemm_bf16<<<gg, 256, 0, stream>>>(Xv, Wvt, bv, 1.f, nullptr, VpT, nullptr,
                                    MTOT, DD, DD);

  flash_attn<<<dim3(SS / 64, HH, BB), 128, 0, stream>>>(Qp, Kp, VpT, kpm, Ctx);

  gemm_bf16<<<gg, 256, 0, stream>>>(Ctx, Wot, bo, 1.f, nullptr, nullptr,
                                    (float*)d_out, MTOT, DD, DD);
  (void)in_sizes; (void)n_in; (void)out_size; (void)ws_size;
}
